// Model_52518860096343
// MI455X (gfx1250) — matrix operand order repaired and checked by reading, not compiled, not run on hardware
//
#include <hip/hip_runtime.h>
#include <stdint.h>

typedef __attribute__((ext_vector_type(16))) _Float16 v16h;
typedef __attribute__((ext_vector_type(8)))  float    v8f;

#define WMMA_F16(A, B, C) \
  __builtin_amdgcn_wmma_f32_16x16x32_f16(false, (A), false, (B), (short)0, (C), false, false)

static constexpr int BATCH = 2;
static constexpr int HEADS = 16;
static constexpr int SEQ   = 2048;
static constexpr int DH    = 64;
static constexpr int BLOCK_M = 64;   // query rows per workgroup (4 waves x 16)
static constexpr int BLOCK_N = 32;   // key tile
static constexpr float SCALE = 0.125f;            // 1/sqrt(64)
static constexpr float LOG2E = 1.4426950408889634f;

// ---------- cross-lane reductions over 16-lane column groups (wave32) ----------
#define SWZ_F(v, imm) \
  __uint_as_float((uint32_t)__builtin_amdgcn_ds_swizzle((int)__float_as_uint(v), (imm)))

__device__ __forceinline__ float red_max16(float v) {
  v = fmaxf(v, SWZ_F(v, 0x041F));  // xor 1
  v = fmaxf(v, SWZ_F(v, 0x081F));  // xor 2
  v = fmaxf(v, SWZ_F(v, 0x101F));  // xor 4
  v = fmaxf(v, SWZ_F(v, 0x201F));  // xor 8
  return v;
}
__device__ __forceinline__ float red_sum16(float v) {
  v += SWZ_F(v, 0x041F);
  v += SWZ_F(v, 0x081F);
  v += SWZ_F(v, 0x101F);
  v += SWZ_F(v, 0x201F);
  return v;
}

// ---------- JAX threefry-2x32, key = (0, 42) ----------
#define TF_ROUND(x0, x1, R) { (x0) += (x1); (x1) = (((x1) << (R)) | ((x1) >> (32 - (R)))); (x1) ^= (x0); }

__device__ __forceinline__ void threefry2x32(uint32_t c0, uint32_t c1,
                                             uint32_t& xo0, uint32_t& xo1) {
  const uint32_t ks0 = 0u, ks1 = 42u;
  const uint32_t ks2 = 0u ^ 42u ^ 0x1BD11BDAu;
  uint32_t x0 = c0 + ks0;
  uint32_t x1 = c1 + ks1;
  TF_ROUND(x0, x1, 13) TF_ROUND(x0, x1, 15) TF_ROUND(x0, x1, 26) TF_ROUND(x0, x1, 6)
  x0 += ks1; x1 += ks2 + 1u;
  TF_ROUND(x0, x1, 17) TF_ROUND(x0, x1, 29) TF_ROUND(x0, x1, 16) TF_ROUND(x0, x1, 24)
  x0 += ks2; x1 += ks0 + 2u;
  TF_ROUND(x0, x1, 13) TF_ROUND(x0, x1, 15) TF_ROUND(x0, x1, 26) TF_ROUND(x0, x1, 6)
  x0 += ks0; x1 += ks1 + 3u;
  TF_ROUND(x0, x1, 17) TF_ROUND(x0, x1, 29) TF_ROUND(x0, x1, 16) TF_ROUND(x0, x1, 24)
  x0 += ks1; x1 += ks2 + 4u;
  TF_ROUND(x0, x1, 13) TF_ROUND(x0, x1, 15) TF_ROUND(x0, x1, 26) TF_ROUND(x0, x1, 6)
  x0 += ks2; x1 += ks0 + 5u;
  xo0 = x0; xo1 = x1;
}

// A-operand loader: lane holds row M=L%16; element e -> K = dbase + 16*half + e
__device__ __forceinline__ v16h load_a_q(const float* __restrict__ row, int half, int dbase) {
  v16h v;
#pragma unroll
  for (int e = 0; e < 16; ++e) {
    const int d = dbase + 16 * half + e;
    v[e] = (_Float16)row[d];
  }
  return v;
}

// B-operand loader for K^T: lane holds key column; element e -> K(d) = dbase + e (16 contiguous f32)
__device__ __forceinline__ v16h load_b_k(const float* __restrict__ row, int dbase) {
  v16h v;
#pragma unroll
  for (int e = 0; e < 16; ++e) v[e] = (_Float16)row[dbase + e];
  return v;
}

__global__ __launch_bounds__(128, 1)
void dual_attn_drop_kernel(const float* __restrict__ q1, const float* __restrict__ q2,
                           const float* __restrict__ k1, const float* __restrict__ k2,
                           const float* __restrict__ v1, const float* __restrict__ v2,
                           float* __restrict__ out) {
  __shared__ _Float16 vT[DH][BLOCK_N];        // (v1+v2) tile, transposed, f16
  __shared__ _Float16 pbuf[4][16][BLOCK_N + 2]; // per-wave P staging (C->A transpose)

  const int tid  = (int)threadIdx.x;
  const int wave = tid >> 5;
  const int lane = tid & 31;
  const int ln16 = lane & 15;
  const int half = lane >> 4;

  const int bh = (int)blockIdx.y;
  const int b  = bh >> 4;        // / HEADS
  const int h  = bh & 15;        // % HEADS
  const int q0 = (int)blockIdx.x * BLOCK_M + wave * 16;

  const size_t base = (size_t)bh * (size_t)(SEQ * DH);
  const float* Q1 = q1 + base; const float* Q2 = q2 + base;
  const float* K1 = k1 + base; const float* K2 = k2 + base;
  const float* V1 = v1 + base; const float* V2 = v2 + base;
  float* O = out + base;

  // resident Q A-operands (f32 -> f16 once)
  const float* qr1 = Q1 + (size_t)(q0 + ln16) * DH;
  const float* qr2 = Q2 + (size_t)(q0 + ln16) * DH;
  const v16h aq1lo = load_a_q(qr1, half, 0);
  const v16h aq1hi = load_a_q(qr1, half, 32);
  const v16h aq2lo = load_a_q(qr2, half, 0);
  const v16h aq2hi = load_a_q(qr2, half, 32);

  v8f oacc[4] = {v8f{}, v8f{}, v8f{}, v8f{}};
  float m[8], l[8];
#pragma unroll
  for (int r = 0; r < 8; ++r) { m[r] = -3.0e38f; l[r] = 0.0f; }

  for (int kb = 0; kb < SEQ; kb += BLOCK_N) {
    __syncthreads();  // protect vT from previous iteration's readers

    // --- stage (v1+v2) tile transposed into LDS as f16 ---
    {
      const int kr = tid >> 2;             // 0..31 key row in tile
      const int dc = (tid & 3) << 4;       // 0,16,32,48
      const float* rv1 = V1 + (size_t)(kb + kr) * DH + dc;
      const float* rv2 = V2 + (size_t)(kb + kr) * DH + dc;
#pragma unroll
      for (int i = 0; i < 16; ++i)
        vT[dc + i][kr] = (_Float16)(rv1[i] + rv2[i]);
    }

    if (kb + BLOCK_N < SEQ) {  // prefetch next tiles -> global_prefetch_b8
      __builtin_prefetch(K1 + (size_t)(kb + BLOCK_N + ln16) * DH, 0, 0);
      __builtin_prefetch(K2 + (size_t)(kb + BLOCK_N + ln16) * DH, 0, 0);
      __builtin_prefetch(V1 + (size_t)(kb + BLOCK_N + lane) * DH, 0, 0);
      __builtin_prefetch(V2 + (size_t)(kb + BLOCK_N + lane) * DH, 0, 0);
    }
    __syncthreads();

    // --- scores: two 16x16 tiles, 8 WMMAs (dual-source, D=64 split in two) ---
    v8f sc[2];
#pragma unroll
    for (int ns = 0; ns < 2; ++ns) {
      const float* rk1 = K1 + (size_t)(kb + ns * 16 + ln16) * DH;
      const float* rk2 = K2 + (size_t)(kb + ns * 16 + ln16) * DH;
      v8f acc{};
      acc = WMMA_F16(aq1lo, load_b_k(rk1, 16 * half),      acc);
      acc = WMMA_F16(aq1hi, load_b_k(rk1, 32 + 16 * half), acc);
      acc = WMMA_F16(aq2lo, load_b_k(rk2, 16 * half),      acc);
      acc = WMMA_F16(aq2hi, load_b_k(rk2, 32 + 16 * half), acc);
      sc[ns] = acc;
    }

    // --- online softmax (row stats replicated across each 16-lane group) ---
    float alpha[8];
#pragma unroll
    for (int r = 0; r < 8; ++r) {
      const float s0 = sc[0][r] * SCALE;
      const float s1 = sc[1][r] * SCALE;
      const float mx = red_max16(fmaxf(s0, s1));
      const float mn = fmaxf(m[r], mx);
      const float a  = exp2f((m[r] - mn) * LOG2E);
      const float p0 = exp2f((s0 - mn) * LOG2E);
      const float p1 = exp2f((s1 - mn) * LOG2E);
      l[r] = l[r] * a + red_sum16(p0 + p1);  // denominator uses pre-dropout p
      m[r] = mn;
      alpha[r] = a;
      sc[0][r] = p0;
      sc[1][r] = p1;
    }
#pragma unroll
    for (int nd = 0; nd < 4; ++nd)
#pragma unroll
      for (int r = 0; r < 8; ++r) oacc[nd][r] = oacc[nd][r] * alpha[r];

    // --- JAX-exact dropout on P, stage dropped P (x2) to LDS in row-major ---
#pragma unroll
    for (int ns = 0; ns < 2; ++ns) {
#pragma unroll
      for (int r = 0; r < 8; ++r) {
        const uint32_t srow = (uint32_t)(q0 + r + 8 * half);
        const uint32_t tcol = (uint32_t)(kb + ns * 16 + ln16);
        const uint32_t hst  = ((uint32_t)h * (uint32_t)SEQ + srow) * (uint32_t)SEQ + tcol;
        uint32_t x0, x1;
        threefry2x32(hst, hst + (1u << 26), x0, x1);
        const uint32_t bits = b ? x1 : x0;
        const float u = __uint_as_float((bits >> 9) | 0x3f800000u) - 1.0f;
        const float pv = (u < 0.5f) ? sc[ns][r] * 2.0f : 0.0f;  // /(1-p)
        pbuf[wave][r + 8 * half][ns * 16 + ln16] = (_Float16)pv;
      }
    }
    __syncthreads();

    // --- P A-operand from LDS; PV with V B-operands from vT: 4 WMMAs ---
    v16h ap;
#pragma unroll
    for (int e = 0; e < 16; ++e) {
      const int kc = 16 * half + e;
      ap[e] = pbuf[wave][ln16][kc];
    }
#pragma unroll
    for (int nd = 0; nd < 4; ++nd) {
      v16h bv;
#pragma unroll
      for (int e = 0; e < 16; ++e) bv[e] = vT[nd * 16 + ln16][16 * half + e];
      oacc[nd] = WMMA_F16(ap, bv, oacc[nd]);
    }
  }

  // --- normalize by pre-dropout softmax sum and store ---
  float inv[8];
#pragma unroll
  for (int r = 0; r < 8; ++r) inv[r] = 1.0f / l[r];
#pragma unroll
  for (int nd = 0; nd < 4; ++nd)
#pragma unroll
    for (int r = 0; r < 8; ++r)
      O[(size_t)(q0 + r + 8 * half) * DH + nd * 16 + ln16] = oacc[nd][r] * inv[r];
}

extern "C" void kernel_launch(void* const* d_in, const int* in_sizes, int n_in,
                              void* d_out, int out_size, void* d_ws, size_t ws_size,
                              hipStream_t stream) {
  (void)in_sizes; (void)n_in; (void)out_size; (void)d_ws; (void)ws_size;
  const float* q1 = (const float*)d_in[0];
  const float* q2 = (const float*)d_in[1];
  const float* k1 = (const float*)d_in[2];
  const float* k2 = (const float*)d_in[3];
  const float* v1 = (const float*)d_in[4];
  const float* v2 = (const float*)d_in[5];
  float* out = (float*)d_out;

  dim3 grid(SEQ / BLOCK_M, BATCH * HEADS);  // (32, 32)
  dim3 block(128);                          // 4 waves (wave32)
  dual_attn_drop_kernel<<<grid, block, 0, stream>>>(q1, q2, k1, k2, v1, v2, out);
}
